// SSDDecoder_20624432956160
// MI455X (gfx1250) — matrix operand order repaired and checked by reading, not compiled, not run on hardware
//
#include <hip/hip_runtime.h>
#include <hip/hip_bf16.h>

typedef __attribute__((ext_vector_type(16))) _Float16 v16h;
typedef __attribute__((ext_vector_type(8)))  float    v8f;

#define BATCH 32
#define NPRI  8732
#define NCLS  81
#define KCAND 256
#define MAXT  200
#define MAXPC 200
#define FLATN (NCLS * KCAND)   // 20736

// ---------------------------------------------------------------------------
// Kernel 1: decode boxes (priors + deltas*variances) and compute foreground
// mask fg[b,n] = (argmax_c probs[b,n,c] != 0)  <=>  max_{c>=1} p[c] > p[0].
// One streaming pass over the 90MB prob tensor warms the 192MB L2 so the
// per-class kernel's strided re-reads hit L2, not HBM.
// ---------------------------------------------------------------------------
__global__ __launch_bounds__(256) void decode_fg_kernel(
    const float* __restrict__ deltas, const float* __restrict__ probs,
    const float* __restrict__ priors, float* __restrict__ boxes,
    int* __restrict__ fg) {
  int t = blockIdx.x * blockDim.x + threadIdx.x;
  if (t >= BATCH * NPRI) return;
  int n = t % NPRI;

  const float4 pr = *(const float4*)(priors + (size_t)n * 4);
  const float4 dl = *(const float4*)(deltas + (size_t)t * 4);
  float d0 = dl.x * 0.1f, d1 = dl.y * 0.1f, d2 = dl.z * 0.2f, d3 = dl.w * 0.2f;
  float ph = pr.z - pr.x, pw = pr.w - pr.y;
  float pcy = pr.x + 0.5f * ph, pcx = pr.y + 0.5f * pw;
  float bh = expf(d2) * ph, bw = expf(d3) * pw;
  float bcy = d0 * ph + pcy, bcx = d1 * pw + pcx;
  float y1 = bcy - 0.5f * bh, x1 = bcx - 0.5f * bw;
  float4 out;
  out.x = fminf(fmaxf(y1, 0.f), 1.f);
  out.y = fminf(fmaxf(x1, 0.f), 1.f);
  out.z = fminf(fmaxf(y1 + bh, 0.f), 1.f);
  out.w = fminf(fmaxf(x1 + bw, 0.f), 1.f);
  *(float4*)(boxes + (size_t)t * 4) = out;

  const float* p = probs + (size_t)t * NCLS;
  float p0 = p[0];
  float m = 0.f;
  for (int c = 1; c < NCLS; ++c) m = fmaxf(m, p[c]);
  fg[t] = (m > p0) ? 1 : 0;
}

// ---------------------------------------------------------------------------
// Shared helper: bitonic sort of 256 (score desc, idx asc) in LDS.
// ---------------------------------------------------------------------------
__device__ __forceinline__ void bitonic256(float* sc, int* ix, int tid) {
  for (int kk = 2; kk <= 256; kk <<= 1) {
    for (int jj = kk >> 1; jj > 0; jj >>= 1) {
      int i = tid, ixj = i ^ jj;
      if (ixj > i) {
        float s1 = sc[i], s2 = sc[ixj];
        int i1 = ix[i], i2 = ix[ixj];
        bool aGreater = (s1 > s2) || (s1 == s2 && i1 < i2);
        bool desc = ((i & kk) == 0);
        if (desc ? !aGreater : aGreater) {
          sc[i] = s2; sc[ixj] = s1;
          ix[i] = i2; ix[ixj] = i1;
        }
      }
      __syncthreads();
    }
  }
}

// ---------------------------------------------------------------------------
// Kernel 2: one block per (b,c). Exact top-256 via 4-pass radix select on
// float bits, bitonic sort, register-resident single-wave greedy NMS, then a
// WMMA-based capped cumsum (tensor-core prefix sum) for the MAX_PER_CLASS cap.
// ---------------------------------------------------------------------------
__global__ __launch_bounds__(256) void per_class_kernel(
    const float* __restrict__ probs, const int* __restrict__ fg,
    const float* __restrict__ boxes, float* __restrict__ cand_score_g,
    float* __restrict__ cand_box_g) {
  __shared__ float     sc[NPRI];
  __shared__ float     cscore[KCAND];
  __shared__ int       cidx[KCAND];
  __shared__ float     cbox[KCAND][4];
  __shared__ unsigned  hist[256];
  __shared__ _Float16  keepf[KCAND];
  __shared__ int       keepflag[KCAND];
  __shared__ unsigned  sh_prefix, sh_needed, gcnt, ecnt;

  const int tid = threadIdx.x;
  const int bc = blockIdx.x;
  const int b = bc / NCLS, c = bc % NCLS;

  // Stage masked class scores into LDS (strided reads, L2-resident).
  const float* pb = probs + (size_t)b * NPRI * NCLS;
  const int* fgb = fg + (size_t)b * NPRI;
  for (int n = tid; n < NPRI; n += 256)
    sc[n] = fgb[n] ? pb[(size_t)n * NCLS + c] : 0.0f;
  if (tid == 0) { sh_prefix = 0u; sh_needed = KCAND; }
  __syncthreads();

  // 4-pass radix select: find bit pattern T of the 256th-largest score.
  for (int pass = 0; pass < 4; ++pass) {
    const int shift = 24 - 8 * pass;
    const unsigned maskKnown = (pass == 0) ? 0u : (0xFFFFFFFFu << (shift + 8));
    hist[tid] = 0u;
    __syncthreads();
    const unsigned pref = sh_prefix;
    for (int n = tid; n < NPRI; n += 256) {
      unsigned u = __float_as_uint(sc[n]);
      if ((u & maskKnown) == pref) atomicAdd(&hist[(u >> shift) & 255u], 1u);
    }
    __syncthreads();
    if (tid == 0) {
      unsigned cnt = 0, need = sh_needed;
      for (int v = 255; v >= 0; --v) {
        unsigned h = hist[v];
        if (cnt + h >= need) {
          sh_prefix = pref | ((unsigned)v << shift);
          sh_needed = need - cnt;
          break;
        }
        cnt += h;
      }
    }
    __syncthreads();
  }
  const unsigned T = sh_prefix;
  const unsigned needEq = sh_needed;
  const unsigned G = KCAND - needEq;

  // Collect the 256 candidates (order canonicalized by the sort below).
  if (tid == 0) { gcnt = 0u; ecnt = 0u; }
  __syncthreads();
  for (int n = tid; n < NPRI; n += 256) {
    unsigned u = __float_as_uint(sc[n]);
    if (u > T) {
      unsigned p = atomicAdd(&gcnt, 1u);
      cscore[p] = sc[n]; cidx[p] = n;
    } else if (u == T) {
      unsigned r = atomicAdd(&ecnt, 1u);
      if (r < needEq) { unsigned p = G + r; cscore[p] = sc[n]; cidx[p] = n; }
    }
  }
  __syncthreads();

  bitonic256(cscore, cidx, tid);

  // Gather candidate boxes.
  {
    int pidx = cidx[tid];
    float4 bx = *(const float4*)(boxes + ((size_t)b * NPRI + pidx) * 4);
    cbox[tid][0] = bx.x; cbox[tid][1] = bx.y;
    cbox[tid][2] = bx.z; cbox[tid][3] = bx.w;
  }
  __syncthreads();

  // Greedy NMS over 256 sorted candidates, held entirely in wave-0 registers.
  if (tid < 32) {
    const int lane = tid;
    float y1[8], x1[8], y2[8], x2[8], ar[8];
    unsigned km = 0u;
#pragma unroll
    for (int s = 0; s < 8; ++s) {
      int j = s * 32 + lane;
      y1[s] = cbox[j][0]; x1[s] = cbox[j][1];
      y2[s] = cbox[j][2]; x2[s] = cbox[j][3];
      ar[s] = (y2[s] - y1[s]) * (x2[s] - x1[s]);
      if (cscore[j] > 0.5f) km |= (1u << s);
    }
#pragma unroll
    for (int si = 0; si < 8; ++si) {
      for (int l = 0; l < 32; ++l) {
        unsigned kms = __shfl(km, l);
        float iy1 = __shfl(y1[si], l), ix1 = __shfl(x1[si], l);
        float iy2 = __shfl(y2[si], l), ix2 = __shfl(x2[si], l);
        if ((kms >> si) & 1u) {       // uniform branch (broadcast value)
          float iar = (iy2 - iy1) * (ix2 - ix1);
#pragma unroll
          for (int s = 0; s < 8; ++s) {
            if (s < si) continue;     // j = s*32+lane <= i : never suppressed
            if ((km >> s) & 1u) {
              bool after = (s > si) || (lane > l);
              if (after) {
                float ih = fminf(y2[s], iy2) - fmaxf(y1[s], iy1);
                float iw = fminf(x2[s], ix2) - fmaxf(x1[s], ix1);
                float inter = fmaxf(ih, 0.f) * fmaxf(iw, 0.f);
                float uni = ar[s] + iar - inter;
                if (inter > 0.5f * fmaxf(uni, 1e-8f)) km &= ~(1u << s);
              }
            }
          }
        }
      }
    }
#pragma unroll
    for (int s = 0; s < 8; ++s)
      keepf[s * 32 + lane] = (_Float16)(((km >> s) & 1u) ? 1.0f : 0.0f);
  }
  __syncthreads();

  // WMMA prefix sum: cumsum(keep) over 256 sorted slots, cap at MAX_PER_CLASS.
  // A (16x32 f16): A[row][K] = keep[16*row+K] for K<16, 0 pad for K>=16.
  // B (32x16 f16): B[k][j] = (k <= j) for k<16 (upper-triangular ones), 0 pad.
  // D[r][j] = inclusive prefix of row r at column j (exact in f32 accum).
  // Executed redundantly by all 8 full waves (EXEC all ones, as WMMA requires).
  {
    const int lane = tid & 31;
    const bool hi = lane >= 16;
    const int row = lane & 15;
    v16h A, Bm;
#pragma unroll
    for (int h = 0; h < 16; ++h) {
      int Kk = (hi ? 8 : 0) + ((h < 8) ? h : (8 + h));   // ISA 16-bit A layout
      A[h]  = (Kk < 16) ? keepf[16 * row + Kk] : (_Float16)0.0f;
      Bm[h] = (Kk < 16 && Kk <= row) ? (_Float16)1.0f : (_Float16)0.0f;
    }
    v8f Cm = {0.f, 0.f, 0.f, 0.f, 0.f, 0.f, 0.f, 0.f};
    Cm = __builtin_amdgcn_wmma_f32_16x16x32_f16(
        false, A, false, Bm, (short)0, Cm, false, false);

    // Cross-row exclusive offsets from row totals D[r][15] (lanes 15 / 31).
    float off_v[8];
    float run = 0.f;
#pragma unroll
    for (int r = 0; r < 16; ++r) {
      float rsv = __shfl(Cm[r & 7], (r < 8) ? 15 : 31);
      if (hi == (r >= 8)) off_v[r & 7] = run;
      run += rsv;
    }
#pragma unroll
    for (int v = 0; v < 8; ++v) {
      int e = hi ? (128 + 16 * v + (lane - 16)) : (16 * v + lane);
      float csum = off_v[v] + Cm[v];
      keepflag[e] = ((float)keepf[e] > 0.5f && csum <= (float)MAXPC) ? 1 : 0;
    }
  }
  __syncthreads();

  // Emit per-class candidates (score zeroed where not kept; boxes always).
  {
    size_t base = (size_t)bc * KCAND;
    bool k = keepflag[tid] != 0;
    cand_score_g[base + tid] = k ? cscore[tid] : 0.0f;
    float4 ob;
    ob.x = cbox[tid][0]; ob.y = cbox[tid][1];
    ob.z = cbox[tid][2]; ob.w = cbox[tid][3];
    *(float4*)(cand_box_g + (base + tid) * 4) = ob;
  }
}

// ---------------------------------------------------------------------------
// Kernel 3: one block per image. Top-200 over 81*256 kept scores (radix
// select + bitonic sort), then write boxes/labels/scores in reference order.
// ---------------------------------------------------------------------------
__global__ __launch_bounds__(256) void per_image_kernel(
    const float* __restrict__ cand_score_g, const float* __restrict__ cand_box_g,
    float* __restrict__ out) {
  __shared__ float    cs2[256];
  __shared__ int      ci2[256];
  __shared__ unsigned hist[256];
  __shared__ unsigned sh_prefix, sh_needed, gcnt, ecnt;

  const int tid = threadIdx.x;
  const int b = blockIdx.x;
  const float* s = cand_score_g + (size_t)b * FLATN;

  if (tid == 0) { sh_prefix = 0u; sh_needed = MAXT; }
  cs2[tid] = -1.0f;
  ci2[tid] = 0x7FFFFFFF;
  __syncthreads();

  for (int pass = 0; pass < 4; ++pass) {
    const int shift = 24 - 8 * pass;
    const unsigned maskKnown = (pass == 0) ? 0u : (0xFFFFFFFFu << (shift + 8));
    hist[tid] = 0u;
    __syncthreads();
    const unsigned pref = sh_prefix;
    for (int n = tid; n < FLATN; n += 256) {
      unsigned u = __float_as_uint(s[n]);
      if ((u & maskKnown) == pref) atomicAdd(&hist[(u >> shift) & 255u], 1u);
    }
    __syncthreads();
    if (tid == 0) {
      unsigned cnt = 0, need = sh_needed;
      for (int v = 255; v >= 0; --v) {
        unsigned h = hist[v];
        if (cnt + h >= need) {
          sh_prefix = pref | ((unsigned)v << shift);
          sh_needed = need - cnt;
          break;
        }
        cnt += h;
      }
    }
    __syncthreads();
  }
  const unsigned T = sh_prefix;
  const unsigned needEq = sh_needed;
  const unsigned G = MAXT - needEq;

  if (tid == 0) { gcnt = 0u; ecnt = 0u; }
  __syncthreads();
  for (int n = tid; n < FLATN; n += 256) {
    unsigned u = __float_as_uint(s[n]);
    if (u > T) {
      unsigned p = atomicAdd(&gcnt, 1u);
      cs2[p] = s[n]; ci2[p] = n;
    } else if (u == T) {
      unsigned r = atomicAdd(&ecnt, 1u);
      if (r < needEq) { unsigned p = G + r; cs2[p] = s[n]; ci2[p] = n; }
    }
  }
  __syncthreads();

  bitonic256(cs2, ci2, tid);

  if (tid < MAXT) {
    float sv = cs2[tid];
    int fi = ci2[tid];
    bool ok = sv > 0.0f;
    float4 bx = make_float4(0.f, 0.f, 0.f, 0.f);
    if (ok) bx = *(const float4*)(cand_box_g + ((size_t)b * FLATN + fi) * 4);
    // Output layout: boxes [B,200,4] | labels [B,200] | scores [B,200]
    float* ob = out + ((size_t)b * MAXT + tid) * 4;
    ob[0] = bx.x; ob[1] = bx.y; ob[2] = bx.z; ob[3] = bx.w;
    out[BATCH * MAXT * 4 + (size_t)b * MAXT + tid] = ok ? (float)(fi >> 8) : 0.0f;
    out[BATCH * MAXT * 5 + (size_t)b * MAXT + tid] = fmaxf(sv, 0.0f);
  }
}

// ---------------------------------------------------------------------------
extern "C" void kernel_launch(void* const* d_in, const int* in_sizes, int n_in,
                              void* d_out, int out_size, void* d_ws, size_t ws_size,
                              hipStream_t stream) {
  (void)in_sizes; (void)n_in; (void)out_size; (void)ws_size;
  const float* deltas = (const float*)d_in[0];   // [B,N,4]
  const float* probs  = (const float*)d_in[1];   // [B,N,C]
  const float* priors = (const float*)d_in[2];   // [N,4]
  float* out = (float*)d_out;

  // Workspace partition (~19MB total).
  float* boxes = (float*)d_ws;                         // B*N*4
  int*   fg    = (int*)(boxes + (size_t)BATCH * NPRI * 4);   // B*N
  float* cand_score = (float*)(fg + (size_t)BATCH * NPRI);   // B*C*K
  float* cand_box   = cand_score + (size_t)BATCH * NCLS * KCAND; // B*C*K*4

  int total = BATCH * NPRI;
  decode_fg_kernel<<<(total + 255) / 256, 256, 0, stream>>>(
      deltas, probs, priors, boxes, fg);
  per_class_kernel<<<BATCH * NCLS, 256, 0, stream>>>(
      probs, fg, boxes, cand_score, cand_box);
  per_image_kernel<<<BATCH, 256, 0, stream>>>(cand_score, cand_box, out);
}
